// TEST_LSTM_48318382080080
// MI455X (gfx1250) — compile-verified, matrix operand order checked by reading, not run on hardware
//
#include <hip/hip_runtime.h>

// ---------------------------------------------------------------------------
// 2-layer LSTM (B=64,T=512,IN=256,H=512) + FC(OUT=128) for gfx1250 (MI455X).
// Persistent fused kernel, 64 WGs (32 per layer, layer1 lags layer0 by one
// step), one grid barrier per timestep. Each WG stages its constant weight
// slice into LDS ONCE (reused for all 512 steps; 320KB LDS/WGP on CDNA5),
// then per step runs v_wmma_f32_16x16x32_bf16 with B fragments from ds_load
// and pipelined A fragments from L2. The layer body is templated on (K, KX)
// so all K-loops have compile-time trip counts and fully unroll (no clamp
// selects; hazard slots filled with independent loads). Cell state lives in
// a v8f register fragment for the whole recurrence.
// ---------------------------------------------------------------------------

typedef __bf16 bf16;
typedef __attribute__((ext_vector_type(16))) __bf16 v16bf;
typedef __attribute__((ext_vector_type(8)))  __bf16 v8bf;
typedef __attribute__((ext_vector_type(8)))  float  v8f;

#define B_    64
#define T_    512
#define IN_   256
#define H_    512
#define G_    2048      // 4*H
#define OUT_  128
#define NWG_L 32        // workgroups per layer (each owns 16 h-columns)
#define NWG   (2 * NWG_L)
#define THREADS 128     // 4 waves -> 4 M-tiles of 16 rows = B=64
#define LDS_PAD 8       // (K+8)/2 % 64 == 4 dwords -> clean bank spread

__device__ __forceinline__ float sigmoidf_(float x) {
    return 1.f / (1.f + __expf(-x));
}

// Load a 16x32 bf16 WMMA fragment from a row-major [rows][ld] array (global
// activations or the LDS weight slice; for weights "row" = local gate column).
// Per-lane layout per CDNA5 ISA (16-bit A 16x32): lanes 0-15 hold K k0+0..7
// and k0+16..23 of row (lane&15); lanes 16-31 hold K k0+8..15 and k0+24..31.
__device__ __forceinline__ v16bf load_frag(const bf16* base,
                                           int ld, int row0, int k0, int lane) {
    const int r  = row0 + (lane & 15);
    const int kk = k0 + ((lane >> 4) << 3);
    const bf16* p = base + (size_t)r * ld + kk;
    v8bf lo = *reinterpret_cast<const v8bf*>(p);       // K = kk   .. kk+7
    v8bf hi = *reinterpret_cast<const v8bf*>(p + 16);  // K = kk+16.. kk+23
    return __builtin_shufflevector(lo, hi,
        0,1,2,3,4,5,6,7,8,9,10,11,12,13,14,15);
}

// Fully-unrolled GEMM over compile-time K region [KBEG, KEND).
// A rows from (abase, ald) at local k = k - KOFF (global memory, prefetched
// one k-step ahead into fresh registers; no prefetch on the last iteration).
// B tiles are the 4 gate groups of the LDS weight slice Wl [64][LDP]
// (local rows g*16+c hold gate g, h-column c).
template<int KBEG, int KEND, int KOFF, int LDP>
__device__ __forceinline__ void gemm_region(
    const bf16* __restrict__ abase, int ald, int row0, int lane,
    const bf16* Wl, v8f& a0, v8f& a1, v8f& a2, v8f& a3) {
    v16bf ac = load_frag(abase, ald, row0, KBEG - KOFF, lane);
#pragma unroll
    for (int k0 = KBEG; k0 < KEND; k0 += 32) {
        v16bf an;
        if (k0 + 32 < KEND)                       // constant-folds per iteration
            an = load_frag(abase, ald, row0, k0 + 32 - KOFF, lane);
        v16bf b0 = load_frag(Wl, LDP,  0, k0, lane);
        v16bf b1 = load_frag(Wl, LDP, 16, k0, lane);
        v16bf b2 = load_frag(Wl, LDP, 32, k0, lane);
        v16bf b3 = load_frag(Wl, LDP, 48, k0, lane);
        a0 = __builtin_amdgcn_wmma_f32_16x16x32_bf16(false, ac, false, b0, (short)0, a0, false, false);
        a1 = __builtin_amdgcn_wmma_f32_16x16x32_bf16(false, ac, false, b1, (short)0, a1, false, false);
        a2 = __builtin_amdgcn_wmma_f32_16x16x32_bf16(false, ac, false, b2, (short)0, a2, false, false);
        a3 = __builtin_amdgcn_wmma_f32_16x16x32_bf16(false, ac, false, b3, (short)0, a3, false, false);
        if (k0 + 32 < KEND)
            ac = an;
    }
}

// Grid-wide barrier: monotonic counter (reset by prep kernel each call).
__device__ __forceinline__ void grid_barrier(unsigned* cnt, int it) {
    __threadfence();
    __syncthreads();
    if (threadIdx.x == 0) {
        __hip_atomic_fetch_add(cnt, 1u, __ATOMIC_RELEASE, __HIP_MEMORY_SCOPE_AGENT);
        const unsigned target = (unsigned)(it + 1) * (unsigned)NWG;
        while (__hip_atomic_load(cnt, __ATOMIC_ACQUIRE, __HIP_MEMORY_SCOPE_AGENT) < target) {
            __builtin_amdgcn_s_sleep(2);
        }
    }
    __syncthreads();
    __threadfence();
}

// One layer's full 512-step recurrence; templated so K-loops fully unroll.
// L1=0: input = xbf (k<KX), recurrent = h1[t-1]; output h1[t].
// L1=1: input = h1[t] (k<KX), recurrent = h2 ping-pong; output h2[(t+1)&1].
template<int K, int KX, int L1>
__device__ __forceinline__ void lstm_layer_run(
    const bf16* __restrict__ W, const float* __restrict__ bs,
    const bf16* __restrict__ xbf, bf16* __restrict__ h1, bf16* __restrict__ h2,
    unsigned* cnt, bf16* smem, int j0, int row0, int lane) {
    constexpr int LDP = K + LDS_PAD;

    // Stage this WG's weight slice into LDS once: local row = g*16 + c maps to
    // global gate column g*H + j0 + c; 16-byte chunks -> ds_store_b128.
    {
        constexpr int CH = K >> 3;                 // 16B chunks per column
        for (int idx = threadIdx.x; idx < 64 * CH; idx += THREADS) {
            const int col  = idx / CH;
            const int k8   = (idx % CH) << 3;
            const int gcol = (col >> 4) * H_ + j0 + (col & 15);
            *reinterpret_cast<v8bf*>(&smem[col * LDP + k8]) =
                *reinterpret_cast<const v8bf*>(&W[(size_t)gcol * K + k8]);
        }
    }
    __syncthreads();

    // Per-lane gate biases (col = g*H + j0 + (lane&15)), constant over steps.
    float bias[4];
#pragma unroll
    for (int g = 0; g < 4; ++g) bias[g] = bs[g * H_ + j0 + (lane & 15)];

    // Cell-state fragment, persistent in registers across all 512 steps.
    v8f c = {0.f, 0.f, 0.f, 0.f, 0.f, 0.f, 0.f, 0.f};

    for (int it = 0; it <= T_; ++it) {
        const int  t      = L1 ? (it - 1) : it;
        const bool active = L1 ? (it >= 1) : (it < T_);
        if (active) {
            v8f a0 = {0.f,0.f,0.f,0.f,0.f,0.f,0.f,0.f};
            v8f a1 = a0, a2 = a0, a3 = a0;
            const int tprev = (t > 0) ? (t - 1) : 0;

            const bf16* ab0; int ld0;
            const bf16* ab1;
            if (L1) { ab0 = h1 + (size_t)t * (B_ * H_);       ld0 = H_;
                      ab1 = h2 + (size_t)(t & 1) * (B_ * H_); }
            else    { ab0 = xbf + (size_t)t * IN_;            ld0 = T_ * IN_;
                      ab1 = h1 + (size_t)tprev * (B_ * H_);   }

            gemm_region<0, KX, 0, LDP>(ab0, ld0, row0, lane, smem, a0, a1, a2, a3);
            if (t > 0)   // t==0: recurrent h is zero -> skip region entirely
                gemm_region<KX, K, KX, LDP>(ab1, H_, row0, lane, smem, a0, a1, a2, a3);

            // Pointwise LSTM update, fully in registers (C/D fragment layout:
            // VGPR r, lanes 0-15 -> M=r, lanes 16-31 -> M=r+8; N = lane&15).
            bf16* hout = L1 ? (h2 + (size_t)((t + 1) & 1) * (B_ * H_))
                            : (h1 + (size_t)t * (B_ * H_));
            const int rr   = row0 + ((lane >> 4) << 3);
            const int colj = j0 + (lane & 15);
#pragma unroll
            for (int r = 0; r < 8; ++r) {
                float iv = sigmoidf_(a0[r] + bias[0]);
                float fv = sigmoidf_(a1[r] + bias[1]);
                float gv = tanhf    (a2[r] + bias[2]);
                float ov = sigmoidf_(a3[r] + bias[3]);
                float cn = fv * c[r] + iv * gv;
                c[r] = cn;
                float hn = ov * tanhf(cn);
                hout[(size_t)(rr + r) * H_ + colj] = (bf16)hn;
            }
        }
        grid_barrier(cnt, it);
    }
}

// ---------------------------------------------------------------------------
// Prep: pack weights to bf16 [gate-col][K] with K = concat(input, recurrent),
// sum biases, convert x to bf16, zero h2 ping-pong, reset barrier counter.
// ---------------------------------------------------------------------------
__global__ void k_prep(const float* __restrict__ Wih0, const float* __restrict__ Whh0,
                       const float* __restrict__ bih0, const float* __restrict__ bhh0,
                       const float* __restrict__ Wih1, const float* __restrict__ Whh1,
                       const float* __restrict__ bih1, const float* __restrict__ bhh1,
                       const float* __restrict__ x,
                       bf16* __restrict__ Wc0, bf16* __restrict__ Wc1,
                       float* __restrict__ bs0, float* __restrict__ bs1,
                       bf16* __restrict__ xbf, bf16* __restrict__ h2,
                       unsigned* __restrict__ cnt) {
    const size_t NW0 = (size_t)G_ * 768;           // [2048][256+512]
    const size_t NW1 = (size_t)G_ * 1024;          // [2048][512+512]
    const size_t NX  = (size_t)B_ * T_ * IN_;
    const size_t NH2 = (size_t)2 * B_ * H_;
    const size_t total = NW0 + NW1 + NX + NH2 + 2 * G_ + 1;
    for (size_t i = (size_t)blockIdx.x * blockDim.x + threadIdx.x; i < total;
         i += (size_t)gridDim.x * blockDim.x) {
        size_t j = i;
        if (j < NW0) {
            size_t gcol = j / 768, k = j % 768;
            float v = (k < IN_) ? Wih0[gcol * IN_ + k] : Whh0[gcol * H_ + (k - IN_)];
            Wc0[j] = (bf16)v;
        } else if ((j -= NW0) < NW1) {
            size_t gcol = j / 1024, k = j % 1024;
            float v = (k < H_) ? Wih1[gcol * H_ + k] : Whh1[gcol * H_ + (k - H_)];
            Wc1[j] = (bf16)v;
        } else if ((j -= NW1) < NX) {
            xbf[j] = (bf16)x[j];
        } else if ((j -= NX) < NH2) {
            h2[j] = (bf16)0.f;
        } else if ((j -= NH2) < 2 * (size_t)G_) {
            if (j < G_) bs0[j]       = bih0[j]       + bhh0[j];
            else        bs1[j - G_]  = bih1[j - G_]  + bhh1[j - G_];
        } else {
            cnt[0] = 0u;
        }
    }
}

// ---------------------------------------------------------------------------
// Persistent fused 2-layer LSTM. WG 0..31: layer0 (step t=it). WG 32..63:
// layer1 (step t=it-1). Each WG owns h-columns [slice*16, slice*16+16); wave w
// owns batch rows [16w, 16w+16).
// ---------------------------------------------------------------------------
__global__ void __launch_bounds__(THREADS, 1)
k_lstm(const bf16* __restrict__ Wc0, const bf16* __restrict__ Wc1,
       const float* __restrict__ bs0, const float* __restrict__ bs1,
       const bf16* __restrict__ xbf, bf16* __restrict__ h1,
       bf16* __restrict__ h2, unsigned* __restrict__ cnt) {
    extern __shared__ __align__(16) bf16 smem[];   // [64][K + LDS_PAD]
    const int wg   = blockIdx.x;
    const int lane = threadIdx.x & 31;
    const int row0 = (threadIdx.x >> 5) * 16;      // wave -> M-tile (batch rows)
    if (wg < NWG_L) {
        lstm_layer_run<768, 256, 0>(Wc0, bs0, xbf, h1, h2, cnt, smem,
                                    wg * 16, row0, lane);
    } else {
        lstm_layer_run<1024, 512, 1>(Wc1, bs1, xbf, h1, h2, cnt, smem,
                                     (wg - NWG_L) * 16, row0, lane);
    }
}

// ---------------------------------------------------------------------------
// Final FC on the last h2 (t=511 result lands in ping-pong buffer 0).
// 64*128*512 MACs: negligible; simple VALU kernel.
// ---------------------------------------------------------------------------
__global__ void k_fc(const bf16* __restrict__ h2last, const float* __restrict__ fcw,
                     const float* __restrict__ fcb, float* __restrict__ out) {
    const int b = blockIdx.x;
    const int o = threadIdx.x;     // 128
    const bf16*  hrow = h2last + (size_t)b * H_;
    const float* wrow = fcw + (size_t)o * H_;
    float acc = fcb[o];
#pragma unroll 8
    for (int k = 0; k < H_; ++k) acc += (float)hrow[k] * wrow[k];
    out[b * OUT_ + o] = acc;
}

extern "C" void kernel_launch(void* const* d_in, const int* in_sizes, int n_in,
                              void* d_out, int out_size, void* d_ws, size_t ws_size,
                              hipStream_t stream) {
    (void)in_sizes; (void)n_in; (void)out_size; (void)ws_size;
    const float* x    = (const float*)d_in[0];
    const float* Wih0 = (const float*)d_in[1];
    const float* Whh0 = (const float*)d_in[2];
    const float* bih0 = (const float*)d_in[3];
    const float* bhh0 = (const float*)d_in[4];
    const float* Wih1 = (const float*)d_in[5];
    const float* Whh1 = (const float*)d_in[6];
    const float* bih1 = (const float*)d_in[7];
    const float* bhh1 = (const float*)d_in[8];
    const float* fcw  = (const float*)d_in[9];
    const float* fcb  = (const float*)d_in[10];

    char* ws = (char*)d_ws;
    size_t off = 0;
    auto take = [&](size_t bytes) -> char* {
        char* p = ws + off;
        off += (bytes + 255) & ~(size_t)255;
        return p;
    };
    bf16*     Wc0 = (bf16*)take((size_t)G_ * 768 * 2);       //  3.0 MB
    bf16*     Wc1 = (bf16*)take((size_t)G_ * 1024 * 2);      //  4.0 MB
    float*    bs0 = (float*)take((size_t)G_ * 4);
    float*    bs1 = (float*)take((size_t)G_ * 4);
    bf16*     xbf = (bf16*)take((size_t)B_ * T_ * IN_ * 2);  // 16.0 MB
    bf16*     h1  = (bf16*)take((size_t)T_ * B_ * H_ * 2);   // 32.0 MB
    bf16*     h2  = (bf16*)take((size_t)2 * B_ * H_ * 2);    // ping-pong
    unsigned* cnt = (unsigned*)take(256);

    // Dynamic LDS: layer1 slice 64 x (1024+8) bf16 = 132,096 B (< 320KB/WG).
    const size_t lds_bytes = (size_t)64 * (1024 + LDS_PAD) * sizeof(bf16);
    (void)hipFuncSetAttribute((const void*)k_lstm,
                              hipFuncAttributeMaxDynamicSharedMemorySize,
                              (int)lds_bytes);

    k_prep<<<2048, 256, 0, stream>>>(Wih0, Whh0, bih0, bhh0,
                                     Wih1, Whh1, bih1, bhh1,
                                     x, Wc0, Wc1, bs0, bs1, xbf, h2, cnt);
    k_lstm<<<NWG, THREADS, lds_bytes, stream>>>(Wc0, Wc1, bs0, bs1, xbf, h1, h2, cnt);
    // Step 511 writes h2 buffer ((511+1)&1)==0 -> final hidden state at h2 + 0.
    k_fc<<<B_, OUT_, 0, stream>>>(h2, fcw, fcb, (float*)d_out);
}
